// PointNetSetAbstraction_47029891891546
// MI455X (gfx1250) — compile-verified, matrix operand order checked by reading, not run on hardware
//
#include <hip/hip_runtime.h>
#include <hip/hip_bf16.h>

typedef __attribute__((ext_vector_type(16))) _Float16 v16h;
typedef __attribute__((ext_vector_type(8)))  _Float16 h8;
typedef __attribute__((ext_vector_type(8)))  float    v8f;

#define NB      16
#define NPTS    8192
#define NPOINT_ 1024
#define NSAMPLE_ 32
#define DCH     64
#define OCH     128
#define RAD2    0.04f
#define PTOT    (NPOINT_ * NSAMPLE_)   // 32768 positions per batch

// Swizzle: channel c -> (bank, slot) matching the 16-bit WMMA operand layout.
// Lane (lr, hi) reads bank==hi; slots [0..15] are K 0..31 (kb=0), [16..31] are K 32..63.
__device__ __forceinline__ int swz_bank(int c) { return (c >> 3) & 1; }
__device__ __forceinline__ int swz_slot(int c) { return (c & 7) + ((c >> 4) << 3); }

// ---------------------------------------------------------------------------
// Kernel 1: furthest point sampling. One block per batch, dists in LDS,
// block-wide argmax tree each iteration. Also writes new_xyz into d_out.
// ---------------------------------------------------------------------------
__global__ __launch_bounds__(256) void fps_kernel(const float* __restrict__ xyz,
                                                  int* __restrict__ fpsidx,
                                                  float* __restrict__ newxyz) {
  __shared__ float dists[NPTS];
  __shared__ float redv[256];
  __shared__ int   redi[256];
  __shared__ int   sfar;
  const int b = blockIdx.x, t = threadIdx.x;
  const float* xs = xyz + (size_t)b * 3 * NPTS;
  const float* ys = xs + NPTS;
  const float* zs = ys + NPTS;
  for (int n = t; n < NPTS; n += 256) dists[n] = 1e10f;
  if (t == 0) sfar = 0;
  __syncthreads();
  for (int it = 0; it < NPOINT_; ++it) {
    const int far = sfar;
    if (t == 0) fpsidx[b * NPOINT_ + it] = far;
    const float cx = xs[far], cy = ys[far], cz = zs[far];
    float best = -1.0f; int bi = 0;
    for (int n = t; n < NPTS; n += 256) {
      float dx = xs[n] - cx, dy = ys[n] - cy, dz = zs[n] - cz;
      float d  = dx * dx + dy * dy + dz * dz;
      float nd = fminf(dists[n], d);
      dists[n] = nd;
      if (nd > best) { best = nd; bi = n; }
    }
    redv[t] = best; redi[t] = bi;
    __syncthreads();
    for (int off = 128; off > 0; off >>= 1) {
      if (t < off) {
        float ov = redv[t + off]; int oi = redi[t + off];
        if (ov > redv[t] || (ov == redv[t] && oi < redi[t])) { redv[t] = ov; redi[t] = oi; }
      }
      __syncthreads();
    }
    if (t == 0) sfar = redi[0];
    __syncthreads();
  }
  for (int s = t; s < NPOINT_; s += 256) {
    int i = fpsidx[b * NPOINT_ + s];
    newxyz[((size_t)b * 3 + 0) * NPOINT_ + s] = xs[i];
    newxyz[((size_t)b * 3 + 1) * NPOINT_ + s] = ys[i];
    newxyz[((size_t)b * 3 + 2) * NPOINT_ + s] = zs[i];
  }
}

// ---------------------------------------------------------------------------
// Kernel 2: ball query, one wave32 per centroid. Ordered append of first 32
// in-range indices via ballot + prefix popcount.
// ---------------------------------------------------------------------------
__global__ __launch_bounds__(128) void ballquery_kernel(const float* __restrict__ xyz,
                                                        const float* __restrict__ newxyz,
                                                        int* __restrict__ idxbuf) {
  const int wave = threadIdx.x >> 5, lane = threadIdx.x & 31;
  const int cg = blockIdx.x * 4 + wave;   // 0..16383
  const int b  = cg >> 10, s = cg & 1023;
  const float* xs = xyz + (size_t)b * 3 * NPTS;
  const float* ys = xs + NPTS;
  const float* zs = ys + NPTS;
  const float cx = newxyz[((size_t)b * 3 + 0) * NPOINT_ + s];
  const float cy = newxyz[((size_t)b * 3 + 1) * NPOINT_ + s];
  const float cz = newxyz[((size_t)b * 3 + 2) * NPOINT_ + s];
  int* outp = idxbuf + (size_t)cg * NSAMPLE_;
  int count = 0, first = 0;
  bool havefirst = false;
  for (int n0 = 0; n0 < NPTS && count < NSAMPLE_; n0 += 32) {
    const int n = n0 + lane;
    float dx = xs[n] - cx, dy = ys[n] - cy, dz = zs[n] - cz;
    bool in = (dx * dx + dy * dy + dz * dz) < RAD2;
    unsigned mask = (unsigned)__ballot(in);
    if (mask) {
      if (!havefirst) { first = n0 + __ffs(mask) - 1; havefirst = true; }
      if (in) {
        int slot = count + __popc(mask & ((1u << lane) - 1u));
        if (slot < NSAMPLE_) outp[slot] = n;
      }
      count += __popc(mask);
    }
  }
  const int pad = havefirst ? first : 0;
  if (count < NSAMPLE_) {
    for (int slot = count + lane; slot < NSAMPLE_; slot += 32) outp[slot] = pad;
  }
}

// ---------------------------------------------------------------------------
// Kernel 3: fused gather + position attention + attn2 + mlp0 + mlp1 (BN folded).
// Activations live in LDS in WMMA-swizzled [pos][bank][32] f16 layout so each
// operand is one contiguous 32B v16h load and each writeback one 16B h8 store.
// Block = 4 centroids (128 positions). Emits swizzled f16 x3 + channel sums.
// ---------------------------------------------------------------------------
__global__ __launch_bounds__(256) void stage1_kernel(
    const float* __restrict__ xyz, const float* __restrict__ points,
    const float* __restrict__ newxyz, const int* __restrict__ idxbuf,
    const float* w0, const float* b0, const float* s0, const float* t0,
    const float* w1, const float* b1, const float* s1, const float* t1,
    const float* w2, const float* b2, const float* s2, const float* t2,
    const float* mw0, const float* mb0, const float* ms0, const float* mt0,
    const float* mw1, const float* mb1, const float* ms1, const float* mt1,
    _Float16* __restrict__ x3, float* __restrict__ meansum) {
  __shared__ alignas(32) _Float16 Wa[64 * 64], Wm0[64 * 64], Wm1[64 * 64]; // swizzled
  __shared__ alignas(32) float Ba[64], Bm0[64], Bm1[64];
  __shared__ float fw0[24], fb0[8], fw1[64 * 8], fb1[64];
  __shared__ alignas(32) _Float16 XA[128 * 64], XB[128 * 64];              // [p][bank][32]
  const int t = threadIdx.x;
  const int b = blockIdx.x >> 8;           // 256 blocks per batch
  const int sbase = (blockIdx.x & 255) * 4;

  // Fold BN into weights/bias while loading, storing weights pre-swizzled.
  for (int e = t; e < 64 * 64; e += 256) {
    int o = e >> 6, c = e & 63;
    int d = (o * 2 + swz_bank(c)) * 32 + swz_slot(c);
    Wa[d]  = (_Float16)(s2[o]  * w2[e]);
    Wm0[d] = (_Float16)(ms0[o] * mw0[e]);
    Wm1[d] = (_Float16)(ms1[o] * mw1[e]);
  }
  if (t < 64) {
    Ba[t]  = s2[t]  * b2[t]  + t2[t];
    Bm0[t] = ms0[t] * mb0[t] + mt0[t];
    Bm1[t] = ms1[t] * mb1[t] + mt1[t];
    fb1[t] = s1[t]  * b1[t]  + t1[t];
  }
  if (t < 24) fw0[t] = s0[t / 3] * w0[t];
  if (t < 8)  fb0[t] = s0[t] * b0[t] + t0[t];
  for (int e = t; e < 64 * 8; e += 256) fw1[e] = s1[e >> 3] * w1[e];
  __syncthreads();

  // Gather grouped points + tiny position-attention branch (VALU), write swizzled.
  {
    const int p = t & 127;                 // position within tile
    const int chalf = t >> 7;              // channel half (0/1)
    const int sl = p >> 5, j = p & 31;
    const int sg = sbase + sl;
    const int pidx = idxbuf[((size_t)b * NPOINT_ + sg) * NSAMPLE_ + j];
    const float* xs = xyz + (size_t)b * 3 * NPTS;
    float gx = xs[pidx]            - newxyz[((size_t)b * 3 + 0) * NPOINT_ + sg];
    float gy = xs[NPTS + pidx]     - newxyz[((size_t)b * 3 + 1) * NPOINT_ + sg];
    float gz = xs[2 * NPTS + pidx] - newxyz[((size_t)b * 3 + 2) * NPOINT_ + sg];
    float h[8];
#pragma unroll
    for (int o = 0; o < 8; ++o) {
      float v = fw0[o * 3 + 0] * gx + fw0[o * 3 + 1] * gy + fw0[o * 3 + 2] * gz + fb0[o];
      h[o] = v > 0.0f ? v : 0.0f;
    }
    const float* pb = points + (size_t)b * DCH * NPTS;
    for (int c = chalf * 32; c < chalf * 32 + 32; ++c) {
      float pos = fb1[c];
#pragma unroll
      for (int o = 0; o < 8; ++o) pos += fw1[c * 8 + o] * h[o];
      XA[(p * 2 + swz_bank(c)) * 32 + swz_slot(c)] =
          (_Float16)(pb[(size_t)c * NPTS + pidx] + pos);
    }
  }
  __syncthreads();

  const int wv = t >> 5, lane = t & 31;
  const int lr = lane & 15, hi = lane >> 4;
  const int pt0 = wv >> 2;                              // 0 or 1, wave-uniform

  auto gemm64 = [&](const _Float16* W, const float* Bias,
                    const _Float16* Xin, _Float16* Xout, bool relu) {
    const int ot = wv & 3;                              // fixed o-tile per wave
    const v16h a0 = *(const v16h*)&W[((ot * 16 + lr) * 2 + hi) * 32];
    const v16h a1 = *(const v16h*)&W[((ot * 16 + lr) * 2 + hi) * 32 + 16];
    const v8f  bias = *(const v8f*)&Bias[ot * 16 + hi * 8];
#pragma unroll
    for (int k = 0; k < 4; ++k) {                       // constant trip count
      const int pt = pt0 + 2 * k;                       // 4 p-tiles per wave
      const _Float16* xp = &Xin[((pt * 16 + lr) * 2 + hi) * 32];
      const v16h b0 = *(const v16h*)xp;
      const v16h b1 = *(const v16h*)(xp + 16);
      v8f acc = bias;
      acc = __builtin_amdgcn_wmma_f32_16x16x32_f16(false, a0, false, b0,
                                                   (short)0, acc, false, false);
      acc = __builtin_amdgcn_wmma_f32_16x16x32_f16(false, a1, false, b1,
                                                   (short)0, acc, false, false);
      h8 res;
#pragma unroll
      for (int r = 0; r < 8; ++r) {
        float v = acc[r];
        if (relu) v = v > 0.0f ? v : 0.0f;
        res[r] = (_Float16)v;
      }
      *(h8*)&Xout[((pt * 16 + lr) * 2 + hi) * 32 + ot * 8] = res;
    }
  };

  gemm64(Wa,  Ba,  XA, XB, false);  // x = bn(attn_w2 @ (points + pos))
  __syncthreads();
  gemm64(Wm0, Bm0, XB, XA, true);   // mlp0 + relu
  __syncthreads();
  gemm64(Wm1, Bm1, XA, XB, true);   // mlp1 + relu
  __syncthreads();

  // Spill x3 keeping the swizzled layout (contiguous -> coalesced b128 stores).
  {
    const uint4* src = (const uint4*)XB;                       // 1024 x 16B
    uint4* dst = (uint4*)(x3 + ((size_t)b * PTOT + (size_t)sbase * NSAMPLE_) * 64);
    for (int e = t; e < 1024; e += 256) dst[e] = src[e];
  }
  // Per-channel partial sums for the ECA mean.
  {
    int c = t & 63, q = t >> 6;
    int bank = swz_bank(c), slot = swz_slot(c);
    float sum = 0.0f;
#pragma unroll
    for (int pp = q * 32; pp < q * 32 + 32; ++pp)
      sum += (float)XB[(pp * 2 + bank) * 32 + slot];
    atomicAdd(&meansum[b * 64 + c], sum);
  }
}

// ---------------------------------------------------------------------------
// Kernel 4: ECA 1D conv (k=3) on channel means + sigmoid -> per (b,c) scale.
// ---------------------------------------------------------------------------
__global__ __launch_bounds__(256) void eca_sig_kernel(const float* __restrict__ meansum,
                                                      const float* __restrict__ eca_k,
                                                      float* __restrict__ sig) {
  const int t = threadIdx.x;
  const float k0 = eca_k[0], k1 = eca_k[1], k2 = eca_k[2];
  const float inv = 1.0f / (float)PTOT;
  for (int e = t; e < NB * 64; e += 256) {
    int c = e & 63;
    float y0 = meansum[e] * inv;
    float ym = (c > 0)  ? meansum[e - 1] * inv : 0.0f;
    float yp = (c < 63) ? meansum[e + 1] * inv : 0.0f;
    float y  = k0 * ym + k1 * y0 + k2 * yp;
    sig[e] = 1.0f / (1.0f + __expf(-y));
  }
}

// ---------------------------------------------------------------------------
// Kernel 5: eca conv (128x64) with sigmoid + BN folded into the A-matrix,
// then max-pool over the 32-sample neighborhood. Block = 2 centroids.
// ---------------------------------------------------------------------------
__global__ __launch_bounds__(256) void stage2_kernel(
    const _Float16* __restrict__ x3, const float* __restrict__ sig,
    const float* __restrict__ ew, const float* __restrict__ eb,
    const float* __restrict__ es, const float* __restrict__ et,
    float* __restrict__ outfeat) {
  __shared__ alignas(32) _Float16 W[128 * 64];      // swizzled
  __shared__ alignas(32) float Bias[128];
  __shared__ alignas(32) _Float16 XA[64 * 64];      // [p][bank][32]
  __shared__ alignas(32) _Float16 XO[64 * 128];     // [p][o]
  const int t = threadIdx.x;
  const int b = blockIdx.x >> 9;            // 512 blocks per batch
  const int sbase = (blockIdx.x & 511) * 2; // 2 centroids -> 64 positions
  for (int e = t; e < 128 * 64; e += 256) {
    int o = e >> 6, c = e & 63;
    W[(o * 2 + swz_bank(c)) * 32 + swz_slot(c)] =
        (_Float16)(es[o] * ew[e] * sig[b * 64 + c]);   // (W diag(s)) x == W (s*x)
  }
  if (t < 128) Bias[t] = es[t] * eb[t] + et[t];
  {
    const uint4* src = (const uint4*)(x3 + ((size_t)b * PTOT + (size_t)sbase * NSAMPLE_) * 64);
    uint4* dst = (uint4*)XA;                          // 512 x 16B, already swizzled
    for (int e = t; e < 512; e += 256) dst[e] = src[e];
  }
  __syncthreads();
  const int wv = t >> 5, lane = t & 31;
  const int lr = lane & 15, hi = lane >> 4;
  {
    const int ot = wv;                                 // 8 o-tiles, one per wave
    const v16h a0 = *(const v16h*)&W[((ot * 16 + lr) * 2 + hi) * 32];
    const v16h a1 = *(const v16h*)&W[((ot * 16 + lr) * 2 + hi) * 32 + 16];
    const v8f  bias = *(const v8f*)&Bias[ot * 16 + hi * 8];
#pragma unroll
    for (int pt = 0; pt < 4; ++pt) {
      const _Float16* xp = &XA[((pt * 16 + lr) * 2 + hi) * 32];
      const v16h b0 = *(const v16h*)xp;
      const v16h b1 = *(const v16h*)(xp + 16);
      v8f acc = bias;
      acc = __builtin_amdgcn_wmma_f32_16x16x32_f16(false, a0, false, b0,
                                                   (short)0, acc, false, false);
      acc = __builtin_amdgcn_wmma_f32_16x16x32_f16(false, a1, false, b1,
                                                   (short)0, acc, false, false);
      h8 res;
#pragma unroll
      for (int r = 0; r < 8; ++r) res[r] = (_Float16)acc[r];
      *(h8*)&XO[(pt * 16 + lr) * 128 + ot * 16 + hi * 8] = res;
    }
  }
  __syncthreads();
  {
    const int o = t >> 1, sl = t & 1;
    float m = -3.4e38f;
#pragma unroll
    for (int j = 0; j < 32; ++j) {
      float v = (float)XO[(sl * 32 + j) * 128 + o];
      m = v > m ? v : m;
    }
    outfeat[((size_t)b * OCH + o) * NPOINT_ + sbase + sl] = m;
  }
}

// ---------------------------------------------------------------------------
extern "C" void kernel_launch(void* const* d_in, const int* in_sizes, int n_in,
                              void* d_out, int out_size, void* d_ws, size_t ws_size,
                              hipStream_t stream) {
  (void)in_sizes; (void)n_in; (void)out_size; (void)ws_size;
  const float* xyz    = (const float*)d_in[0];
  const float* points = (const float*)d_in[1];
  const float* aw0 = (const float*)d_in[2],  *ab0 = (const float*)d_in[3];
  const float* as0 = (const float*)d_in[4],  *at0 = (const float*)d_in[5];
  const float* aw1 = (const float*)d_in[6],  *ab1 = (const float*)d_in[7];
  const float* as1 = (const float*)d_in[8],  *at1 = (const float*)d_in[9];
  const float* aw2 = (const float*)d_in[10], *ab2 = (const float*)d_in[11];
  const float* as2 = (const float*)d_in[12], *at2 = (const float*)d_in[13];
  const float* mw0 = (const float*)d_in[14], *mb0 = (const float*)d_in[15];
  const float* ms0 = (const float*)d_in[16], *mt0 = (const float*)d_in[17];
  const float* mw1 = (const float*)d_in[18], *mb1 = (const float*)d_in[19];
  const float* ms1 = (const float*)d_in[20], *mt1 = (const float*)d_in[21];
  const float* ecak = (const float*)d_in[22];
  const float* ecaw = (const float*)d_in[23], *ecab = (const float*)d_in[24];
  const float* ecas = (const float*)d_in[25], *ecat = (const float*)d_in[26];

  char* ws = (char*)d_ws;
  int*       fpsidx  = (int*)ws;                                  // 64 KB
  int*       ballidx = (int*)(ws + (64u << 10));                  // 2 MB
  float*     meansum = (float*)(ws + (64u << 10) + (2u << 20));   // 4 KB
  float*     sig     = meansum + NB * 64;                         // 4 KB
  _Float16*  x3      = (_Float16*)(ws + (64u << 10) + (2u << 20) + (8u << 10)); // 64 MB

  float* newxyz  = (float*)d_out;                       // (16,3,1024)
  float* outfeat = newxyz + (size_t)NB * 3 * NPOINT_;   // (16,128,1024)

  hipMemsetAsync(meansum, 0, NB * 64 * sizeof(float), stream);
  fps_kernel<<<NB, 256, 0, stream>>>(xyz, fpsidx, newxyz);
  ballquery_kernel<<<NB * NPOINT_ / 4, 128, 0, stream>>>(xyz, newxyz, ballidx);
  stage1_kernel<<<NB * 256, 256, 0, stream>>>(
      xyz, points, newxyz, ballidx,
      aw0, ab0, as0, at0, aw1, ab1, as1, at1, aw2, ab2, as2, at2,
      mw0, mb0, ms0, mt0, mw1, mb1, ms1, mt1, x3, meansum);
  eca_sig_kernel<<<1, 256, 0, stream>>>(meansum, ecak, sig);
  stage2_kernel<<<NB * 512, 256, 0, stream>>>(x3, sig, ecaw, ecab, ecas, ecat, outfeat);
}
